// ImitateJoint_87857851007057
// MI455X (gfx1250) — compile-verified, matrix operand order checked by reading, not run on hardware
//
#include <hip/hip_runtime.h>
#include <hip/hip_bf16.h>
#include <math.h>
#include <stdint.h>

// ---------------- problem constants (from reference) ----------------
static constexpr int cB   = 64;    // images
static constexpr int cK   = 8;     // beam width
static constexpr int cT   = 12;    // program length
static constexpr int cH   = 2048;  // hidden
static constexpr int cIN  = 2048;  // CNN feature
static constexpr int cD   = 400;   // grammar size
static constexpr int cE   = 128;   // op embedding
static constexpr int cBK  = cB * cK;            // 512
static constexpr int cKIH = 2 * cIN + cE;       // 4224 (W_ih K)
static constexpr int cKCAT= cKIH + cH;          // 6272 (fused K: [xf|emb|xf3|h])
static constexpr int c4H  = 4 * cH;             // 8192

typedef __bf16 bf16;
typedef __attribute__((ext_vector_type(16))) __bf16 v16bf;
typedef __attribute__((ext_vector_type(8)))  __bf16 v8bf;
typedef __attribute__((ext_vector_type(8)))  float  v8f;

union FragBF { v16bf v; v8bf h[2]; };

// ---- CDNA5 async global->LDS copy (ASYNCcnt path), guarded fallback ----
#if defined(__HIP_DEVICE_COMPILE__) && __has_builtin(__builtin_amdgcn_global_load_async_to_lds_b128)
#define ASYNC_LDS 1
#else
#define ASYNC_LDS 0
#endif

#if ASYNC_LDS
// builtin signature (from hipcc diagnostic): pointee is a 16B int vector,
// global operand in addrspace(1), LDS operand in addrspace(3)
typedef int v4i_t __attribute__((vector_size(4 * sizeof(int))));
typedef __attribute__((address_space(1))) v4i_t* g_v4i_p;
typedef __attribute__((address_space(3))) v4i_t* l_v4i_p;
#endif

__device__ __forceinline__ void async_cp16(const bf16* g, bf16* l)
{
#if ASYNC_LDS
    // flat LDS address carries the LDS offset in its low 32 bits (ISA aperture rule);
    // flat global address == addrspace(1) address.
    __builtin_amdgcn_global_load_async_to_lds_b128(
        (g_v4i_p)(uintptr_t)g,
        (l_v4i_p)(uint32_t)(uintptr_t)l,
        0, 0);
#else
    *(v8bf*)l = *(const v8bf*)g;   // sync staging fallback (global_load + ds_store)
#endif
}

__device__ __forceinline__ void wait_async_all()
{
#if ASYNC_LDS
#if __has_builtin(__builtin_amdgcn_s_wait_asynccnt)
    __builtin_amdgcn_s_wait_asynccnt(0);
#else
    asm volatile("s_wait_asynccnt 0x0" ::: "memory");
#endif
#endif
}

// =====================================================================
// bf16 WMMA GEMM:  Out[M,N] = A[M,Kd] * B[N,Kd]^T (+bias / fused epilogue)
// Block tile 64x128, 8 waves; each wave owns one 16-wide N strip and 4
// 16x16 accum tiles; K chained in 32-deep v_wmma_f32_16x16x32_bf16 steps.
// Double-buffered LDS, tiles streamed with GLOBAL_LOAD_ASYNC_TO_LDS_B128
// (one s_wait_asynccnt + one barrier per K chunk).
// MODE 0: f32 out = acc + bias          (LSTM pre-activations z)
// MODE 1: bf16 out = relu(acc + bias)   (fc1 -> hd)
// MODE 2: f32 out = acc + bias + mask_table[pop[m]][n]  (logits)
// =====================================================================
template <int MODE>
__global__ __launch_bounds__(256)
void gemm_bf16(const bf16* __restrict__ A, int lda,
               const bf16* __restrict__ Bw, int ldb,
               const float* __restrict__ bias,
               void* __restrict__ Out, int M, int N, int Kd,
               const float* __restrict__ mask_table,
               const int* __restrict__ pop_t)
{
    __shared__ bf16 sA[2][64 * 40];    // 32 cols + 8 pad, double buffered
    __shared__ bf16 sB[2][128 * 40];

    const int tid  = threadIdx.x;
    const int lane = tid & 31;
    const int wave = tid >> 5;
    const int half = lane >> 4;     // K-half select per CDNA5 layout
    const int mn   = lane & 15;     // row (A) / col (B) within 16
    const int bm   = blockIdx.y * 64;
    const int bn   = blockIdx.x * 128;

    v8f acc[4];
#pragma unroll
    for (int ms = 0; ms < 4; ++ms)
#pragma unroll
        for (int i = 0; i < 8; ++i) acc[ms][i] = 0.0f;

    const int ar = tid >> 2;          // 0..63
    const int ac = (tid & 3) * 8;     // 0,8,16,24
    const int br = tid >> 1;          // 0..127
    const int bc = (tid & 1) * 16;    // 0,16

    const bool bOK = (bn + br) < N;

    // zero-fill ragged-N rows of both buffers once; async loads never touch them
    if (!bOK) {
        v8bf z;
#pragma unroll
        for (int i = 0; i < 8; ++i) z[i] = (bf16)0.0f;
        *(v8bf*)(&sB[0][br * 40 + bc])     = z;
        *(v8bf*)(&sB[0][br * 40 + bc + 8]) = z;
        *(v8bf*)(&sB[1][br * 40 + bc])     = z;
        *(v8bf*)(&sB[1][br * 40 + bc + 8]) = z;
    }

    const bf16* gA = A  + (size_t)(bm + ar) * lda + ac;
    const bf16* gB = Bw + (size_t)(bn + br) * ldb + bc;

    // stage first K chunk into buffer 0
    async_cp16(gA, &sA[0][ar * 40 + ac]);
    if (bOK) {
        async_cp16(gB,     &sB[0][br * 40 + bc]);
        async_cp16(gB + 8, &sB[0][br * 40 + bc + 8]);
    }

    for (int kb = 0; kb < Kd; kb += 32) {
        const int buf = (kb >> 5) & 1;
        wait_async_all();     // tile for this chunk landed in LDS
        __syncthreads();      // visible to all waves; prior buffer reads done

        const int kn = kb + 32;
        if (kn < Kd) {        // stream next chunk into the other buffer
            const int nb = buf ^ 1;
            async_cp16(gA + kn, &sA[nb][ar * 40 + ac]);
            if (bOK) {
                async_cp16(gB + kn,     &sB[nb][br * 40 + bc]);
                async_cp16(gB + kn + 8, &sB[nb][br * 40 + bc + 8]);
                if (kn + 32 < Kd)   // keep the L2 weight stream warm
                    __builtin_prefetch(gB + kn + 32, 0, 1);
            }
        }

        // CDNA5 16-bit fragment layout: lane holds K = {h8..h8+7} u {16+h8..}
        const int k0 = half * 8;
        FragBF bfrag;
        bfrag.h[0] = *(const v8bf*)(&sB[buf][(wave * 16 + mn) * 40 + k0]);
        bfrag.h[1] = *(const v8bf*)(&sB[buf][(wave * 16 + mn) * 40 + 16 + k0]);
#pragma unroll
        for (int ms = 0; ms < 4; ++ms) {
            FragBF afrag;
            afrag.h[0] = *(const v8bf*)(&sA[buf][(ms * 16 + mn) * 40 + k0]);
            afrag.h[1] = *(const v8bf*)(&sA[buf][(ms * 16 + mn) * 40 + 16 + k0]);
            acc[ms] = __builtin_amdgcn_wmma_f32_16x16x32_bf16(
                false, afrag.v, false, bfrag.v, (short)0, acc[ms], false, false);
        }
    }

    // ---- epilogue: lane mn = n, vgpr i -> m = half*8 + i ----
    const int n = bn + wave * 16 + mn;
    if (n < N) {
        const float bv = bias ? bias[n] : 0.0f;
#pragma unroll
        for (int ms = 0; ms < 4; ++ms) {
#pragma unroll
            for (int i = 0; i < 8; ++i) {
                const int m = bm + ms * 16 + half * 8 + i;
                float v = acc[ms][i] + bv;
                if (MODE == 0) {
                    ((float*)Out)[(size_t)m * N + n] = v;
                } else if (MODE == 1) {
                    ((bf16*)Out)[(size_t)m * N + n] = (bf16)fmaxf(v, 0.0f);
                } else {
                    v += mask_table[(size_t)pop_t[m] * cD + n];
                    ((float*)Out)[(size_t)m * N + n] = v;
                }
            }
        }
    }
}

// ---------------- one-time per-launch prep ----------------
__global__ void k_pack_wz(const float* __restrict__ Wih,
                          const float* __restrict__ Whh,
                          bf16* __restrict__ Wz)
{
    size_t i = (size_t)blockIdx.x * blockDim.x + threadIdx.x;
    const size_t total = (size_t)c4H * cKCAT;
    if (i >= total) return;
    const int r = (int)(i / cKCAT);
    const int c = (int)(i % cKCAT);
    const float v = (c < cKIH) ? Wih[(size_t)r * cKIH + c]
                               : Whh[(size_t)r * cH + (c - cKIH)];
    Wz[i] = (bf16)v;
}

__global__ void k_cvt_bf16(const float* __restrict__ src, bf16* __restrict__ dst, int n)
{
    int i = blockIdx.x * blockDim.x + threadIdx.x;
    if (i < n) dst[i] = (bf16)src[i];
}

__global__ void k_bias_add(const float* __restrict__ a, const float* __restrict__ b,
                           float* __restrict__ o, int n)
{
    int i = blockIdx.x * blockDim.x + threadIdx.x;
    if (i < n) o[i] = a[i] + b[i];
}

// Xcat layout per row bk: [xf(2048) | emb(128) | xf3(2048) | h(2048)]
__global__ void k_init_xcat(const float* __restrict__ x_f, bf16* __restrict__ Xcat)
{
    int i = blockIdx.x * blockDim.x + threadIdx.x;
    if (i >= cBK * cKCAT) return;
    const int bk = i / cKCAT, c = i % cKCAT;
    float v = 0.0f;
    if (c < cIN) v = x_f[(bk % cB) * cIN + c];  // xf = tile(x_f,(K,1))
    Xcat[i] = (bf16)v;
}

// fill per-step emb (relu(W_emb[:,pop]+b_emb)) and xf3 columns of Xcat
__global__ void k_emb_xf3(const int* __restrict__ pop_t,
                          const float* __restrict__ W_emb,
                          const float* __restrict__ b_emb,
                          const float* __restrict__ xf3_t,
                          bf16* __restrict__ Xcat)
{
    int i = blockIdx.x * blockDim.x + threadIdx.x;  // over 512*(E+IN)
    if (i >= cBK * (cE + cIN)) return;
    const int bk = i / (cE + cIN), c = i % (cE + cIN);
    if (c < cE) {
        const int p = pop_t[bk];
        const float v = fmaxf(W_emb[c * cD + p] + b_emb[c], 0.0f);
        Xcat[(size_t)bk * cKCAT + cIN + c] = (bf16)v;
    } else {
        const int cc = c - cE;
        Xcat[(size_t)bk * cKCAT + cIN + cE + cc] = (bf16)xf3_t[(size_t)bk * cIN + cc];
    }
}

// LSTM gates: z cols [i|f|g|o], c2 = sig(f)*c + sig(i)*tanh(g), h2 = sig(o)*tanh(c2)
__global__ void k_gates(const float* __restrict__ z, const float* __restrict__ c_cur,
                        float* __restrict__ c2, float* __restrict__ h2,
                        bf16* __restrict__ hbb)
{
    int i = blockIdx.x * blockDim.x + threadIdx.x;  // 512*2048
    if (i >= cBK * cH) return;
    const int bk = i >> 11, j = i & (cH - 1);
    const float* zr = z + (size_t)bk * c4H;
    const float ig = zr[j], fg = zr[cH + j], gg = zr[2 * cH + j], og = zr[3 * cH + j];
    const float si = 1.0f / (1.0f + expf(-ig));
    const float sf = 1.0f / (1.0f + expf(-fg));
    const float so = 1.0f / (1.0f + expf(-og));
    const float cn = sf * c_cur[i] + si * tanhf(gg);
    const float hn = so * tanhf(cn);
    c2[i] = cn;
    h2[i] = hn;
    hbb[i] = (bf16)hn;
}

// row log_softmax in place + argmax (greedy)
__global__ __launch_bounds__(128)
void k_softmax(float* __restrict__ lt, int* __restrict__ greedy)
{
    const int bk = blockIdx.x, tid = threadIdx.x;
    __shared__ float sv[128];
    __shared__ int   si[128];
    float* row = lt + (size_t)bk * cD;
    float mx = -INFINITY; int mi = 0x7FFFFFFF;
    for (int d = tid; d < cD; d += 128) {
        const float x = row[d];
        if (x > mx || (x == mx && d < mi)) { mx = x; mi = d; }
    }
    sv[tid] = mx; si[tid] = mi; __syncthreads();
    for (int s = 64; s > 0; s >>= 1) {
        if (tid < s) {
            if (sv[tid + s] > sv[tid] ||
                (sv[tid + s] == sv[tid] && si[tid + s] < si[tid])) {
                sv[tid] = sv[tid + s]; si[tid] = si[tid + s];
            }
        }
        __syncthreads();
    }
    const float rmax = sv[0];
    if (tid == 0) greedy[bk] = si[0];
    __syncthreads();
    float acc = 0.0f;
    for (int d = tid; d < cD; d += 128) acc += expf(row[d] - rmax);
    sv[tid] = acc; __syncthreads();
    for (int s = 64; s > 0; s >>= 1) { if (tid < s) sv[tid] += sv[tid + s]; __syncthreads(); }
    const float lse = rmax + logf(sv[0]);
    for (int d = tid; d < cD; d += 128) row[d] -= lse;
}

// Gumbel-with-maximum: g per candidate
__global__ __launch_bounds__(128)
void k_gphi(const float* __restrict__ lt, const float* __restrict__ logp,
            const float* __restrict__ G, const float* __restrict__ u_t,
            float* __restrict__ gmat)
{
    const int bk = blockIdx.x, tid = threadIdx.x;
    __shared__ float sv[128];
    __shared__ float sphi[cD];
    const float* lrow = lt + (size_t)bk * cD;
    const float* urow = u_t + (size_t)bk * cD;
    const float lp = logp[bk], Gv = G[bk];
    float mx = -INFINITY;
    for (int d = tid; d < cD; d += 128) {
        const float uu = fminf(fmaxf(urow[d], 1e-9f), 1.0f - 1e-9f);
        const float gum = -logf(-logf(uu));
        const float gp = lrow[d] + lp + gum;
        sphi[d] = gp;
        mx = fmaxf(mx, gp);
    }
    sv[tid] = mx; __syncthreads();
    for (int s = 64; s > 0; s >>= 1) { if (tid < s) sv[tid] = fmaxf(sv[tid], sv[tid + s]); __syncthreads(); }
    const float Z = sv[0];
    __syncthreads();
    for (int d = tid; d < cD; d += 128) {
        const float gp = sphi[d];
        const float v  = Gv - gp + log1pf(-expf(gp - Z));
        const float g  = Gv - fmaxf(v, 0.0f) - log1pf(expf(-fabsf(v)));
        gmat[(size_t)bk * cD + d] = g;
    }
}

// per-image top-8 over K*D candidates (iterative masked argmax)
__global__ __launch_bounds__(256)
void k_topk(const float* __restrict__ gmat, float* __restrict__ gval, int* __restrict__ gidx)
{
    const int b = blockIdx.x, tid = threadIdx.x;
    __shared__ float sv[256];
    __shared__ int   si[256];
    __shared__ int   chosen[cK];
    for (int kk = 0; kk < cK; ++kk) {
        float best = -INFINITY; int bi = 0x7FFFFFFF;
        for (int j = tid; j < cK * cD; j += 256) {
            bool skip = false;
            for (int q = 0; q < kk; ++q) if (chosen[q] == j) skip = true;
            if (skip) continue;
            const int k = j / cD, d = j - k * cD;
            const float v = gmat[(size_t)(k * cB + b) * cD + d];
            if (v > best || (v == best && j < bi)) { best = v; bi = j; }
        }
        sv[tid] = best; si[tid] = bi; __syncthreads();
        for (int s = 128; s > 0; s >>= 1) {
            if (tid < s) {
                if (sv[tid + s] > sv[tid] ||
                    (sv[tid + s] == sv[tid] && si[tid + s] < si[tid])) {
                    sv[tid] = sv[tid + s]; si[tid] = si[tid + s];
                }
            }
            __syncthreads();
        }
        if (tid == 0) { gval[b * cK + kk] = sv[0]; gidx[b * cK + kk] = si[0]; chosen[kk] = si[0]; }
        __syncthreads();
    }
}

// beam select: order/sample, carry logp/G/samples/outs (ping-pong)
__global__ void k_select(int t, const float* __restrict__ lt, const int* __restrict__ greedy,
                         const float* __restrict__ gval, const int* __restrict__ gidx,
                         const float* __restrict__ logp_c, const float* __restrict__ G_c,
                         const float* __restrict__ smp_c, const float* __restrict__ outs_c,
                         float* __restrict__ logp_n, float* __restrict__ G_n,
                         float* __restrict__ smp_n, float* __restrict__ outs_n,
                         int* __restrict__ order)
{
    const int bk = blockIdx.x * blockDim.x + threadIdx.x;
    if (bk >= cBK) return;
    const int k = bk / cB, b = bk % cB;   // bk = k*B + b
    int ord, smp; float Gn;
    if (t == 0) {
        ord = bk; smp = greedy[bk]; Gn = G_c[bk];
    } else {
        const int j = gidx[b * cK + k];
        ord = (j / cD) * cB + b; smp = j % cD; Gn = gval[b * cK + k];
    }
    const float sel = lt[(size_t)ord * cD + smp];
    order[bk]  = ord;
    logp_n[bk] = (t == 0) ? logp_c[ord] : logp_c[ord] + sel;
    G_n[bk]    = Gn;
    for (int tt = 0; tt < cT; ++tt) {
        smp_n[bk * cT + tt]  = smp_c[ord * cT + tt];
        outs_n[bk * cT + tt] = outs_c[ord * cT + tt];
    }
    smp_n[bk * cT + t]  = (float)smp;
    outs_n[bk * cT + t] = sel;
}

// ent_full[bk] = sum_d lp*exp(lp)*mask_bin over gathered rows
__global__ __launch_bounds__(128)
void k_entfull(const float* __restrict__ lt, const int* __restrict__ order,
               const int* __restrict__ pop_t, const float* __restrict__ mask_table,
               float* __restrict__ entfull)
{
    const int bk = blockIdx.x, tid = threadIdx.x;
    __shared__ float sv[128];
    const int ord = order[bk];
    const int p = pop_t[ord];
    const float* row = lt + (size_t)ord * cD;
    const float* mrow = mask_table + (size_t)p * cD;
    float acc = 0.0f;
    for (int d = tid; d < cD; d += 128) {
        const float lp = row[d];
        acc += (mrow[d] == 0.0f) ? lp * expf(lp) : 0.0f;
    }
    sv[tid] = acc; __syncthreads();
    for (int s = 64; s > 0; s >>= 1) { if (tid < s) sv[tid] += sv[tid + s]; __syncthreads(); }
    if (tid == 0) entfull[bk] = sv[0];
}

// gather h2/c2 rows by order; h goes straight into Xcat (bf16 h-columns)
__global__ void k_gather_hc(const int* __restrict__ order, const float* __restrict__ h2,
                            const float* __restrict__ c2, float* __restrict__ c_next,
                            bf16* __restrict__ Xcat)
{
    int i = blockIdx.x * blockDim.x + threadIdx.x;  // 512*2048
    if (i >= cBK * cH) return;
    const int bk = i >> 11, j = i & (cH - 1);
    const int ord = order[bk];
    c_next[i] = c2[(size_t)ord * cH + j];
    Xcat[(size_t)bk * cKCAT + cKIH + j] = (bf16)h2[(size_t)ord * cH + j];
}

__device__ __forceinline__ float gls(float x)  // log P(Gumbel > x)
{
    const float y = expf(-x);
    if (x >= 10.0f) return -x - 0.5f * y + y * y * (1.0f / 24.0f);
    return logf(-expm1f(-expf(-fminf(x, 10.0f))));
}

// importance-weighted entropy update (t > 0)
__global__ void k_entropy(const float* __restrict__ gval, const float* __restrict__ logp_n,
                          const float* __restrict__ entfull, float* __restrict__ ent,
                          float* __restrict__ wpq, float* __restrict__ lgp)
{
    const int b = blockIdx.x * blockDim.x + threadIdx.x;
    if (b >= cB) return;
    const float gk = gval[b * cK + (cK - 1)];
    float sw = 0.0f, upd = 0.0f;
    for (int k = 0; k < cK - 1; ++k) {
        const float phik = logp_n[k * cB + b];
        const float w = expf(phik - gls(gk - phik));
        upd += w * entfull[k * cB + b];
        sw  += w;
        wpq[b * (cK - 1) + k] = w;
        lgp[b * (cK - 1) + k] = phik;
    }
    ent[b] += upd / sw;
}

// final layout: outs_take | samples_take | ent | wpq | lgp  (all f32)
__global__ void k_epilogue(const float* __restrict__ outs_f, const float* __restrict__ smp_f,
                           const float* __restrict__ ent, const float* __restrict__ wpq,
                           const float* __restrict__ lgp, float* __restrict__ out)
{
    const int i = blockIdx.x * blockDim.x + threadIdx.x;
    const int n1 = cB * (cK - 1) * cT;   // 5376
    if (i < n1) {
        const int b = i / ((cK - 1) * cT);
        const int r = i % ((cK - 1) * cT);
        const int k = r / cT, tt = r % cT;
        out[i]      = outs_f[(k * cB + b) * cT + tt];
        out[n1 + i] = smp_f[(k * cB + b) * cT + tt];
    } else if (i < n1 + cB) {
        out[2 * n1 + (i - n1)] = ent[i - n1];
    } else if (i < n1 + cB + cB * (cK - 1)) {
        const int q = i - n1 - cB;
        out[2 * n1 + cB + q] = wpq[q];
        out[2 * n1 + cB + cB * (cK - 1) + q] = lgp[q];
    }
}

// =====================================================================
extern "C" void kernel_launch(void* const* d_in, const int* in_sizes, int n_in,
                              void* d_out, int out_size, void* d_ws, size_t ws_size,
                              hipStream_t stream)
{
    const float* x_f        = (const float*)d_in[0];
    const float* x_f3_seq   = (const float*)d_in[1];
    const float* gumbel_u   = (const float*)d_in[2];
    const int*   pop_syms   = (const int*)  d_in[3];
    const float* mask_table = (const float*)d_in[4];
    const float* W_emb      = (const float*)d_in[5];
    const float* b_emb      = (const float*)d_in[6];
    const float* W_ih       = (const float*)d_in[7];
    const float* W_hh       = (const float*)d_in[8];
    const float* b_ih       = (const float*)d_in[9];
    const float* b_hh       = (const float*)d_in[10];
    const float* W_fc1      = (const float*)d_in[11];
    const float* b_fc1      = (const float*)d_in[12];
    const float* W_out      = (const float*)d_in[13];
    const float* b_out      = (const float*)d_in[14];

    char* ws = (char*)d_ws;
    size_t off = 0;
    auto carve = [&](size_t bytes) -> void* {
        void* p = ws + off;
        off += (bytes + 255) & ~(size_t)255;
        return p;
    };

    bf16*  Wz     = (bf16*) carve((size_t)c4H * cKCAT * 2);   // [W_ih|W_hh] bf16
    bf16*  Wfc1b  = (bf16*) carve((size_t)cH * cH * 2);
    bf16*  Woutb  = (bf16*) carve((size_t)cD * cH * 2);
    float* biasz  = (float*)carve((size_t)c4H * 4);
    bf16*  Xcat   = (bf16*) carve((size_t)cBK * cKCAT * 2);
    float* zbuf   = (float*)carve((size_t)cBK * c4H * 4);
    float* cbuf0  = (float*)carve((size_t)cBK * cH * 4);
    float* cbuf1  = (float*)carve((size_t)cBK * cH * 4);
    float* c2b    = (float*)carve((size_t)cBK * cH * 4);
    float* h2b    = (float*)carve((size_t)cBK * cH * 4);
    bf16*  hbb    = (bf16*) carve((size_t)cBK * cH * 2);
    bf16*  hd     = (bf16*) carve((size_t)cBK * cH * 2);
    float* lt     = (float*)carve((size_t)cBK * cD * 4);      // logits -> logp_tok
    float* gmat   = (float*)carve((size_t)cBK * cD * 4);
    int*   greedy = (int*)  carve((size_t)cBK * 4);
    float* logp0  = (float*)carve((size_t)cBK * 4);
    float* logp1  = (float*)carve((size_t)cBK * 4);
    float* G0     = (float*)carve((size_t)cBK * 4);
    float* G1     = (float*)carve((size_t)cBK * 4);
    float* smp0   = (float*)carve((size_t)cBK * cT * 4);
    float* smp1   = (float*)carve((size_t)cBK * cT * 4);
    float* outs0  = (float*)carve((size_t)cBK * cT * 4);
    float* outs1  = (float*)carve((size_t)cBK * cT * 4);
    int*   order  = (int*)  carve((size_t)cBK * 4);
    float* gval   = (float*)carve((size_t)cB * cK * 4);
    int*   gidx   = (int*)  carve((size_t)cB * cK * 4);
    float* entful = (float*)carve((size_t)cBK * 4);
    float* ent    = (float*)carve((size_t)cB * 4);
    float* wpq    = (float*)carve((size_t)cB * (cK - 1) * 4);
    float* lgp    = (float*)carve((size_t)cB * (cK - 1) * 4);

    float* logpP[2] = {logp0, logp1};
    float* GP[2]    = {G0, G1};
    float* smpP[2]  = {smp0, smp1};
    float* outsP[2] = {outs0, outs1};
    float* cP[2]    = {cbuf0, cbuf1};

    // ---- one-time prep: pack weights to bf16 (fits in 192MB L2 across T steps) ----
    {
        const size_t tot = (size_t)c4H * cKCAT;
        k_pack_wz<<<(unsigned)((tot + 255) / 256), 256, 0, stream>>>(W_ih, W_hh, Wz);
    }
    { const int n = cH * cH; k_cvt_bf16<<<(n + 255) / 256, 256, 0, stream>>>(W_fc1, Wfc1b, n); }
    { const int n = cD * cH; k_cvt_bf16<<<(n + 255) / 256, 256, 0, stream>>>(W_out, Woutb, n); }
    k_bias_add<<<(c4H + 255) / 256, 256, 0, stream>>>(b_ih, b_hh, biasz, c4H);
    { const int n = cBK * cKCAT; k_init_xcat<<<(n + 255) / 256, 256, 0, stream>>>(x_f, Xcat); }
    (void)hipMemsetAsync(cbuf0, 0, (size_t)cBK * cH * 4, stream);
    (void)hipMemsetAsync(logp0, 0, (size_t)cBK * 4, stream);
    (void)hipMemsetAsync(G0,    0, (size_t)cBK * 4, stream);
    (void)hipMemsetAsync(smp0,  0, (size_t)cBK * cT * 4, stream);
    (void)hipMemsetAsync(outs0, 0, (size_t)cBK * cT * 4, stream);
    (void)hipMemsetAsync(ent,   0, (size_t)cB * 4, stream);

    // ---- scan over T steps ----
    for (int t = 0; t < cT; ++t) {
        const int cur = t & 1, nxt = cur ^ 1;
        const int*   pop_t = pop_syms + t * cBK;
        const float* xf3_t = x_f3_seq + (size_t)t * cBK * cIN;
        const float* u_t   = gumbel_u + (size_t)t * cBK * cD;

        { const int n = cBK * (cE + cIN);
          k_emb_xf3<<<(n + 255) / 256, 256, 0, stream>>>(pop_t, W_emb, b_emb, xf3_t, Xcat); }

        { dim3 g(c4H / 128, cBK / 64);            // fused LSTM GEMM (bulk of FLOPs)
          gemm_bf16<0><<<g, 256, 0, stream>>>(Xcat, cKCAT, Wz, cKCAT, biasz,
                                              zbuf, cBK, c4H, cKCAT, nullptr, nullptr); }
        { const int n = cBK * cH;
          k_gates<<<(n + 255) / 256, 256, 0, stream>>>(zbuf, cP[cur], c2b, h2b, hbb); }
        { dim3 g(cH / 128, cBK / 64);             // fc1 + relu -> bf16
          gemm_bf16<1><<<g, 256, 0, stream>>>(hbb, cH, Wfc1b, cH, b_fc1,
                                              hd, cBK, cH, cH, nullptr, nullptr); }
        { dim3 g((cD + 127) / 128, cBK / 64);     // logits + mask gather
          gemm_bf16<2><<<g, 256, 0, stream>>>(hd, cH, Woutb, cH, b_out,
                                              lt, cBK, cD, cH, mask_table, pop_t); }

        k_softmax<<<cBK, 128, 0, stream>>>(lt, greedy);
        if (t > 0) {
            k_gphi<<<cBK, 128, 0, stream>>>(lt, logpP[cur], GP[cur], u_t, gmat);
            k_topk<<<cB, 256, 0, stream>>>(gmat, gval, gidx);
        }
        k_select<<<(cBK + 255) / 256, 256, 0, stream>>>(
            t, lt, greedy, gval, gidx,
            logpP[cur], GP[cur], smpP[cur], outsP[cur],
            logpP[nxt], GP[nxt], smpP[nxt], outsP[nxt], order);
        if (t > 0) {
            k_entfull<<<cBK, 128, 0, stream>>>(lt, order, pop_t, mask_table, entful);
            k_entropy<<<1, cB, 0, stream>>>(gval, logpP[nxt], entful, ent, wpq, lgp);
        }
        { const int n = cBK * cH;
          k_gather_hc<<<(n + 255) / 256, 256, 0, stream>>>(order, h2b, c2b, cP[nxt], Xcat); }
    }

    // after an even number of steps the live carries are in buffer index 0
    {
        const int total = cB * (cK - 1) * cT + cB + 2 * cB * (cK - 1);  // 6336 threads, 5888 active
        k_epilogue<<<(total + 255) / 256, 256, 0, stream>>>(outsP[0], smpP[0], ent, wpq, lgp,
                                                            (float*)d_out);
    }
}
